// UniCL_7791070675491
// MI455X (gfx1250) — matrix operand order repaired and checked by reading, not compiled, not run on hardware
//
#include <hip/hip_runtime.h>
#include <hip/hip_bf16.h>

// ---------------------------------------------------------------------------
// UniCL loss, collapsed form:
//   loss = (S2 - 2*s*S1) / (2*B*L)
//   S1 = sum_{j,d} u[j,d]*v[j,d],  u = mask^T @ img, v = mask^T @ txt
//   S2 = sum_i rw[i]*(rowLSE[i] + colLSE[i]),  rw[i] = sum_j mask[i,j]*cm[j]
//   rowLSE/colLSE: logsumexp of rows/cols of s*(img @ txt^T)   [the only big op]
// GEMM done with V_WMMA_F32_16X16X32_BF16, f32 inputs converted to bf16
// in-register; exp-sum epilogue fused (fixed offset for range safety).
// ---------------------------------------------------------------------------

typedef __attribute__((ext_vector_type(16))) __bf16 v16bf;
typedef __attribute__((ext_vector_type(8)))  float  v8f;

#define BDIM 768
#define NB   4096
#define LLAB 32
#define LSE_OFF 20.0f

// A fragment (16x32 bf16, M on lanes, K packed in VGPRs):
//   f[e<8]  = mat[row0 + m][kb + 8h + e]
//   f[e>=8] = mat[row0 + m][kb + 16 + 8h + (e-8)]
__device__ __forceinline__ v16bf load_fragA(const float* __restrict__ mat,
                                            int row0, int kb, int m, int h) {
  const float* r = mat + (size_t)(row0 + m) * BDIM + kb + 8 * h;
  float4 a0 = *(const float4*)(r + 0);
  float4 a1 = *(const float4*)(r + 4);
  float4 b0 = *(const float4*)(r + 16);
  float4 b1 = *(const float4*)(r + 20);
  v16bf f;
  f[0]=(__bf16)a0.x; f[1]=(__bf16)a0.y; f[2]=(__bf16)a0.z; f[3]=(__bf16)a0.w;
  f[4]=(__bf16)a1.x; f[5]=(__bf16)a1.y; f[6]=(__bf16)a1.z; f[7]=(__bf16)a1.w;
  f[8]=(__bf16)b0.x; f[9]=(__bf16)b0.y; f[10]=(__bf16)b0.z; f[11]=(__bf16)b0.w;
  f[12]=(__bf16)b1.x; f[13]=(__bf16)b1.y; f[14]=(__bf16)b1.z; f[15]=(__bf16)b1.w;
  return f;
}

// B fragment (32x16 bf16, N on lanes, K rows striped across lane halves):
//   f[e<8]  = txt[col0 + n][kb + 8h + e]
//   f[e>=8] = txt[col0 + n][kb + 16 + 8h + (e-8)]
__device__ __forceinline__ v16bf load_fragB(const float* __restrict__ mat,
                                            int col0, int kb, int n, int h) {
  const float* r = mat + (size_t)(col0 + n) * BDIM + kb + 8 * h;
  float4 c0 = *(const float4*)(r + 0);
  float4 c1 = *(const float4*)(r + 4);
  float4 c2 = *(const float4*)(r + 16);
  float4 c3 = *(const float4*)(r + 20);
  v16bf f;
  f[0]=(__bf16)c0.x;  f[1]=(__bf16)c0.y;  f[2]=(__bf16)c0.z;  f[3]=(__bf16)c0.w;
  f[4]=(__bf16)c1.x;  f[5]=(__bf16)c1.y;  f[6]=(__bf16)c1.z;  f[7]=(__bf16)c1.w;
  f[8]=(__bf16)c2.x;  f[9]=(__bf16)c2.y;  f[10]=(__bf16)c2.z; f[11]=(__bf16)c2.w;
  f[12]=(__bf16)c3.x; f[13]=(__bf16)c3.y; f[14]=(__bf16)c3.z; f[15]=(__bf16)c3.w;
  return f;
}

// 256 threads = 8 wave32. Wave (wy,wx) computes 32 rows x 64 cols (2x4 WMMA
// tiles). Block = 128x128. Grid = 32x32. Fused exp-sum epilogue.
__global__ __launch_bounds__(256) void gemm_lse_kernel(
    const float* __restrict__ img, const float* __restrict__ txt,
    float* __restrict__ row_sum, float* __restrict__ col_sum) {
  const int lane = threadIdx.x & 31;
  const int wid  = threadIdx.x >> 5;
  const int wy   = wid >> 1;
  const int wx   = wid & 1;
  const int m    = lane & 15;   // row within A tile / col within B,C tile
  const int h    = lane >> 4;   // lane half
  const int row0 = blockIdx.y * 128 + wy * 32;
  const int col0 = blockIdx.x * 128 + wx * 64;

  v8f acc[2][4];
#pragma unroll
  for (int rt = 0; rt < 2; ++rt)
#pragma unroll
    for (int t = 0; t < 4; ++t)
#pragma unroll
      for (int r = 0; r < 8; ++r) acc[rt][t][r] = 0.0f;

  for (int kb = 0; kb < BDIM; kb += 32) {
    v16bf a0 = load_fragA(img, row0,      kb, m, h);
    v16bf a1 = load_fragA(img, row0 + 16, kb, m, h);
    v16bf b0 = load_fragB(txt, col0 +  0, kb, m, h);
    v16bf b1 = load_fragB(txt, col0 + 16, kb, m, h);
    v16bf b2 = load_fragB(txt, col0 + 32, kb, m, h);
    v16bf b3 = load_fragB(txt, col0 + 48, kb, m, h);
    acc[0][0] = __builtin_amdgcn_wmma_f32_16x16x32_bf16(false, a0, false, b0, (short)0, acc[0][0], false, false);
    acc[0][1] = __builtin_amdgcn_wmma_f32_16x16x32_bf16(false, a0, false, b1, (short)0, acc[0][1], false, false);
    acc[0][2] = __builtin_amdgcn_wmma_f32_16x16x32_bf16(false, a0, false, b2, (short)0, acc[0][2], false, false);
    acc[0][3] = __builtin_amdgcn_wmma_f32_16x16x32_bf16(false, a0, false, b3, (short)0, acc[0][3], false, false);
    acc[1][0] = __builtin_amdgcn_wmma_f32_16x16x32_bf16(false, a1, false, b0, (short)0, acc[1][0], false, false);
    acc[1][1] = __builtin_amdgcn_wmma_f32_16x16x32_bf16(false, a1, false, b1, (short)0, acc[1][1], false, false);
    acc[1][2] = __builtin_amdgcn_wmma_f32_16x16x32_bf16(false, a1, false, b2, (short)0, acc[1][2], false, false);
    acc[1][3] = __builtin_amdgcn_wmma_f32_16x16x32_bf16(false, a1, false, b3, (short)0, acc[1][3], false, false);
  }

  // Fused epilogue: e = exp(s*c - OFF); reduce rows across lanes 0..15 of each
  // half, cols across the two halves; atomically accumulate into global sums.
  const float s = __logf(20.0f) * 0.17677669529663687f;  // log(1/T)/sqrt(L)
  float rowp[2][8];
  float colp[4] = {0.f, 0.f, 0.f, 0.f};
#pragma unroll
  for (int rt = 0; rt < 2; ++rt)
#pragma unroll
    for (int r = 0; r < 8; ++r) rowp[rt][r] = 0.f;

#pragma unroll
  for (int rt = 0; rt < 2; ++rt)
#pragma unroll
    for (int t = 0; t < 4; ++t)
#pragma unroll
      for (int r = 0; r < 8; ++r) {
        float e = __expf(s * acc[rt][t][r] - LSE_OFF);
        rowp[rt][r] += e;
        colp[t] += e;
      }

#pragma unroll
  for (int rt = 0; rt < 2; ++rt)
#pragma unroll
    for (int r = 0; r < 8; ++r) {
      float x = rowp[rt][r];
      x += __shfl_xor(x, 1);
      x += __shfl_xor(x, 2);
      x += __shfl_xor(x, 4);
      x += __shfl_xor(x, 8);
      rowp[rt][r] = x;  // lanes in each 16-lane half now hold the row total
    }
  if (m == 0) {
#pragma unroll
    for (int rt = 0; rt < 2; ++rt)
#pragma unroll
      for (int r = 0; r < 8; ++r)
        atomicAdd(&row_sum[row0 + rt * 16 + 8 * h + r], rowp[rt][r]);
  }
#pragma unroll
  for (int t = 0; t < 4; ++t) {
    float x = colp[t] + __shfl_xor(colp[t], 16);
    if (h == 0) atomicAdd(&col_sum[col0 + 16 * t + m], x);
  }
}

// u[j,d] = sum_i mask[i,j]*feat[i,d].  grid = (L, 2): y==0 -> img->u, y==1 -> txt->v
__global__ __launch_bounds__(256) void masked_colsum_kernel(
    const float* __restrict__ img, const float* __restrict__ txt,
    const int* __restrict__ labels, float* __restrict__ u, float* __restrict__ v) {
  const int j = blockIdx.x;
  const float* src = (blockIdx.y == 0) ? img : txt;
  float* dst = (blockIdx.y == 0) ? u : v;
  const int d = threadIdx.x;
  float a0 = 0.f, a1 = 0.f, a2 = 0.f;
  for (int i = 0; i < NB; ++i) {
    if (labels[i * LLAB + j] != 0) {          // uniform across the block
      const float* r = src + (size_t)i * BDIM;
      a0 += r[d];
      a1 += r[d + 256];
      a2 += r[d + 512];
    }
  }
  dst[j * BDIM + d]       = a0;
  dst[j * BDIM + d + 256] = a1;
  dst[j * BDIM + d + 512] = a2;
}

// Single block: cm[j], S1 = <u,v>, S2 = sum_i rw[i]*(rowLSE+colLSE), write loss.
__global__ __launch_bounds__(256) void finalize_kernel(
    const int* __restrict__ labels, const float* __restrict__ row_sum,
    const float* __restrict__ col_sum, const float* __restrict__ u,
    const float* __restrict__ v, float* __restrict__ out) {
  __shared__ float cms[LLAB];
  __shared__ float red1[256];
  __shared__ float red2[256];
  __shared__ float S1sh;
  const int t = threadIdx.x;
  if (t < LLAB) cms[t] = 0.f;
  __syncthreads();
  {
    const int j = t & 31;
    const int seg = t >> 5;  // 8 segments of 512 rows
    float c = 0.f;
    for (int i = seg * 512; i < (seg + 1) * 512; ++i)
      c += (labels[i * LLAB + j] != 0) ? 1.f : 0.f;
    atomicAdd(&cms[j], c);
  }
  __syncthreads();

  float s1 = 0.f;
  for (int idx = t; idx < LLAB * BDIM; idx += 256) s1 += u[idx] * v[idx];

  float s2 = 0.f;
  for (int i = t; i < NB; i += 256) {
    float rw = 0.f;
#pragma unroll
    for (int j = 0; j < LLAB; ++j)
      rw += (labels[i * LLAB + j] != 0) ? cms[j] : 0.f;
    float lse = __logf(row_sum[i]) + __logf(col_sum[i]) + 2.0f * LSE_OFF;
    s2 += rw * lse;
  }

  red1[t] = s1;
  __syncthreads();
  for (int o = 128; o > 0; o >>= 1) {
    if (t < o) red1[t] += red1[t + o];
    __syncthreads();
  }
  if (t == 0) S1sh = red1[0];
  __syncthreads();
  red2[t] = s2;
  __syncthreads();
  for (int o = 128; o > 0; o >>= 1) {
    if (t < o) red2[t] += red2[t + o];
    __syncthreads();
  }
  if (t == 0) {
    const float s = __logf(20.0f) * 0.17677669529663687f;
    const float S1 = S1sh;
    const float S2 = red2[0];
    out[0] = (S2 - 2.0f * s * S1) / (2.0f * (float)NB * (float)LLAB);
  }
}

extern "C" void kernel_launch(void* const* d_in, const int* in_sizes, int n_in,
                              void* d_out, int out_size, void* d_ws, size_t ws_size,
                              hipStream_t stream) {
  (void)in_sizes; (void)n_in; (void)out_size; (void)ws_size;
  const float* img    = (const float*)d_in[0];
  const float* txt    = (const float*)d_in[1];
  const int*   labels = (const int*)d_in[2];
  float* out = (float*)d_out;

  float* ws      = (float*)d_ws;
  float* row_sum = ws;                       // 4096
  float* col_sum = ws + NB;                  // 4096
  float* u       = ws + 2 * NB;              // 32*768
  float* v       = ws + 2 * NB + LLAB*BDIM;  // 32*768

  hipMemsetAsync(row_sum, 0, 2 * NB * sizeof(float), stream);

  dim3 gg(NB / 128, NB / 128);
  gemm_lse_kernel<<<gg, 256, 0, stream>>>(img, txt, row_sum, col_sum);

  dim3 gc(LLAB, 2);
  masked_colsum_kernel<<<gc, 256, 0, stream>>>(img, txt, labels, u, v);

  finalize_kernel<<<1, 256, 0, stream>>>(labels, row_sum, col_sum, u, v, out);
}
